// QLSTM_44633300140399
// MI455X (gfx1250) — 8-bit matrix operand order repaired and checked by reading, compiled, not run on hardware
//
#include <hip/hip_runtime.h>
#include <stdint.h>


typedef __attribute__((ext_vector_type(8))) int v8i;

#define B_  64
#define S_  512
#define V_  32000
#define E_  512
#define H_  512
#define G_  2048
#define O_  2
#define NWG_ 64


__device__ __forceinline__ float p2scale(unsigned maxbits) {
    float m = __uint_as_float(maxbits);
    if (!(m > 0.f)) return 1.f;
    return exp2f(ceilf(log2f(m * (1.0f / 127.0f))));
}

__device__ __forceinline__ int clampi(int v, int lo, int hi) {
    return v < lo ? lo : (v > hi ? hi : v);
}

__device__ __forceinline__ void grid_barrier(unsigned* cnt, unsigned* gen, unsigned nwg) {
    __syncthreads();
    if (threadIdx.x == 0) {
        __threadfence();
        volatile unsigned* vg = gen;
        unsigned g = *vg;
        unsigned a = atomicAdd(cnt, 1u);
        if (a == nwg - 1u) {
            atomicExch(cnt, 0u);
            __threadfence();
            atomicAdd(gen, 1u);
        } else {
            while (*vg == g) { __builtin_amdgcn_s_sleep(2); }
        }
        __threadfence();
    }
    __syncthreads();
}


__global__ void k_init(unsigned* hdr) {
    if (threadIdx.x < 16) hdr[threadIdx.x] = 0u;
}

__global__ void k_maxabs(const float* __restrict__ w, int n, unsigned* __restrict__ slot) {
    __shared__ float red[256];
    float m = 0.f;
    for (int i = blockIdx.x * blockDim.x + threadIdx.x; i < n; i += gridDim.x * blockDim.x)
        m = fmaxf(m, fabsf(w[i]));
    red[threadIdx.x] = m;
    __syncthreads();
    for (int s = 128; s > 0; s >>= 1) {
        if ((int)threadIdx.x < s) red[threadIdx.x] = fmaxf(red[threadIdx.x], red[threadIdx.x + s]);
        __syncthreads();
    }
    if (threadIdx.x == 0) atomicMax(slot, __float_as_uint(red[0]));
}

__global__ void k_quantw(const float* __restrict__ w, int8_t* __restrict__ q, int n,
                         const unsigned* __restrict__ slot) {
    const float inv = 1.f / p2scale(*slot);
    for (int i = blockIdx.x * blockDim.x + threadIdx.x; i < n; i += gridDim.x * blockDim.x)
        q[i] = (int8_t)clampi((int)rintf(w[i] * inv), -128, 127);
}

__global__ void k_bias(const float* __restrict__ bih, const float* __restrict__ bhh,
                       float* __restrict__ b) {
    int i = blockIdx.x * blockDim.x + threadIdx.x;
    if (i < G_) b[i] = bih[i] + bhh[i];
}

__global__ void k_embq(const int* __restrict__ text, const float* __restrict__ emb,
                       int8_t* __restrict__ xq, const unsigned* __restrict__ slot) {
    const float s   = p2scale(*slot);
    const float inv = 1.f / s;
    const int total = B_ * S_ * E_;
    for (int i = blockIdx.x * blockDim.x + threadIdx.x; i < total; i += gridDim.x * blockDim.x) {
        const int e  = i & (E_ - 1);
        const int bs = i >> 9;
        const int tok = text[bs];
        const float w  = emb[(size_t)tok * E_ + e];
        const float w8 = (float)clampi((int)rintf(w * inv), -128, 127);
        xq[i] = (int8_t)clampi((int)rintf(w8 * s * 16.0f), -128, 127);
    }
}


__device__ __forceinline__ v8i load_a_frag(const int8_t* __restrict__ row, int base) {
    const uint2 q0 = *(const uint2*)(row + base);
    const uint2 q1 = *(const uint2*)(row + base + 16);
    const uint2 q2 = *(const uint2*)(row + base + 32);
    const uint2 q3 = *(const uint2*)(row + base + 48);
    v8i a = {(int)q0.x, (int)q0.y, (int)q1.x, (int)q1.y,
             (int)q2.x, (int)q2.y, (int)q3.x, (int)q3.y};
    return a;
}

__device__ __forceinline__ v8i load_b_frag(const int8_t* __restrict__ row, int kb) {
    const uint2 p0 = *(const uint2*)(row + kb);
    const uint2 p1 = *(const uint2*)(row + kb + 16);
    const uint2 p2 = *(const uint2*)(row + kb + 32);
    const uint2 p3 = *(const uint2*)(row + kb + 48);
    v8i b = {(int)p0.x, (int)p0.y, (int)p1.x, (int)p1.y,
             (int)p2.x, (int)p2.y, (int)p3.x, (int)p3.y};
    return b;
}


__device__ __forceinline__ void cell_update(const int8_t* __restrict__ gbuf,
                                            int8_t* __restrict__ hq, int u, float& c) {
    const int b = u >> 9, j = u & 511;
    const int8_t* gb = gbuf + (size_t)b * G_;
    const float gi = (float)gb[j]          * 0.125f;
    const float gf = (float)gb[j + H_]     * 0.125f;
    const float gg = (float)gb[j + 2 * H_] * 0.125f;
    const float go = (float)gb[j + 3 * H_] * 0.125f;
    const float i_ = fminf(fmaxf(rintf(256.f / (1.f + __expf(-gi))), 0.f), 255.f) * (1.f / 256.f);
    const float f_ = fminf(fmaxf(rintf(256.f / (1.f + __expf(-gf))), 0.f), 255.f) * (1.f / 256.f);
    const float o_ = fminf(fmaxf(rintf(256.f / (1.f + __expf(-go))), 0.f), 255.f) * (1.f / 256.f);
    const float g_ = (float)clampi((int)rintf(tanhf(gg) * 128.f), -128, 127) * (1.f / 128.f);
    c = f_ * c + i_ * g_;
    const float tq = (float)clampi((int)rintf(tanhf(c) * 128.f), -128, 127) * (1.f / 128.f);
    hq[u] = (int8_t)clampi((int)rintf(o_ * tq * 16.f), -128, 127);
}


__global__ void __launch_bounds__(256, 1)
k_lstm(const int8_t* __restrict__ xq, const int8_t* __restrict__ wih,
       const int8_t* __restrict__ whh, const float* __restrict__ bias,
       const unsigned* __restrict__ hdr, int8_t* __restrict__ gbuf,
       int8_t* __restrict__ hq, unsigned* __restrict__ bar) {
    const int lane = threadIdx.x & 31;
    const int wave = threadIdx.x >> 5;
    const int W    = blockIdx.x * 8 + wave;
    const int m0   = (W >> 7) << 4;
    const int n0   = (W & 127) << 4;

    const float s_ih = p2scale(hdr[1]);
    const float s_hh = p2scale(hdr[2]);
    const float fx = 0.0625f * s_ih;
    const float fh = 0.0625f * s_hh;

    const int arow  = m0 + (lane & 15);
    const int abase = (lane < 16) ? 0 : 8;
    const int col   = n0 + (lane & 15);
    const int bbase = (lane < 16) ? 0 : 8;
    const float bcol = bias[col];

    const int8_t* __restrict__ wirow = wih + (size_t)col * E_;
    const int8_t* __restrict__ whrow = whh + (size_t)col * H_;
    const int8_t* __restrict__ hrow  = hq  + (size_t)arow * H_;
    const int8_t* __restrict__ xbase = xq + (size_t)arow * S_ * E_;

    const int tid = blockIdx.x * 256 + threadIdx.x;
    const int u0 = tid, u1 = tid + 16384;
    float c0 = 0.f, c1 = 0.f;

    hq[u0] = 0; hq[u1] = 0;
    grid_barrier(bar, bar + 1, NWG_);

    for (int t = 0; t < S_; ++t) {
        v8i accx0 = {0, 0, 0, 0, 0, 0, 0, 0};
        v8i accx1 = {0, 0, 0, 0, 0, 0, 0, 0};
        v8i acch0 = {0, 0, 0, 0, 0, 0, 0, 0};
        v8i acch1 = {0, 0, 0, 0, 0, 0, 0, 0};
        const int8_t* __restrict__ xrow = xbase + (size_t)t * E_;

        if (t + 1 < S_) {
            const int8_t* nx = xrow + E_;
            __builtin_prefetch(nx, 0, 1);
            __builtin_prefetch(nx + 256, 0, 1);
        }

#pragma unroll
        for (int kc = 0; kc < 8; kc += 2) {
            const int ka0 = kc * 64 + abase;
            const int kb0 = kc * 64 + bbase;
            const int ka1 = ka0 + 64;
            const int kb1 = kb0 + 64;
            const v8i ax0 = load_a_frag(xrow, ka0);
            const v8i ax1 = load_a_frag(xrow, ka1);
            const v8i ah0 = load_a_frag(hrow, ka0);
            const v8i ah1 = load_a_frag(hrow, ka1);
            const v8i bx0 = load_b_frag(wirow, kb0);
            const v8i bx1 = load_b_frag(wirow, kb1);
            const v8i bh0 = load_b_frag(whrow, kb0);
            const v8i bh1 = load_b_frag(whrow, kb1);
            accx0 = __builtin_amdgcn_wmma_i32_16x16x64_iu8(true, ax0, true, bx0, accx0,
                                                           false, false);
            acch0 = __builtin_amdgcn_wmma_i32_16x16x64_iu8(true, ah0, true, bh0, acch0,
                                                           false, false);
            accx1 = __builtin_amdgcn_wmma_i32_16x16x64_iu8(true, ax1, true, bx1, accx1,
                                                           false, false);
            acch1 = __builtin_amdgcn_wmma_i32_16x16x64_iu8(true, ah1, true, bh1, acch1,
                                                           false, false);
        }
        const v8i accx = accx0 + accx1;
        const v8i acch = acch0 + acch1;

#pragma unroll
        for (int r = 0; r < 8; ++r) {
            const int brow = m0 + r + ((lane < 16) ? 0 : 8);
            const float g = (float)accx[r] * fx + (float)acch[r] * fh + bcol;
            gbuf[(size_t)brow * G_ + col] = (int8_t)clampi((int)rintf(g * 8.f), -128, 127);
        }
        grid_barrier(bar, bar + 1, NWG_);

        cell_update(gbuf, hq, u0, c0);
        cell_update(gbuf, hq, u1, c1);
        grid_barrier(bar, bar + 1, NWG_);
    }
}


__global__ void k_fc(const int8_t* __restrict__ hq, const float* __restrict__ fcw,
                     const unsigned* __restrict__ hdr, float* __restrict__ out) {
    const int t = threadIdx.x;
    if (t >= B_ * O_) return;
    const int b = t >> 1, o = t & 1;
    const float s = p2scale(hdr[3]);
    const float inv = 1.f / s;
    float acc = 0.f;
    for (int j = 0; j < H_; ++j) {
        const float wq = (float)clampi((int)rintf(fcw[o * H_ + j] * inv), -128, 127) * s;
        acc += ((float)hq[b * H_ + j] * 0.0625f) * wq;
    }
    out[t] = acc;
}


extern "C" void kernel_launch(void* const* d_in, const int* in_sizes, int n_in,
                              void* d_out, int out_size, void* d_ws, size_t ws_size,
                              hipStream_t stream) {
    (void)in_sizes; (void)n_in; (void)out_size; (void)ws_size;
    const int*   text = (const int*)d_in[0];
    const float* emb  = (const float*)d_in[2];
    const float* w_ih = (const float*)d_in[3];
    const float* w_hh = (const float*)d_in[4];
    const float* b_ih = (const float*)d_in[5];
    const float* b_hh = (const float*)d_in[6];
    const float* fc_w = (const float*)d_in[7];
    float* out = (float*)d_out;

    uint8_t* ws   = (uint8_t*)d_ws;
    unsigned* hdr = (unsigned*)ws;
    float*  bias  = (float*)(ws + 256);
    int8_t* wih_q = (int8_t*)(ws + 8448);
    int8_t* whh_q = wih_q + (size_t)G_ * E_;
    int8_t* xq    = whh_q + (size_t)G_ * H_;
    int8_t* gbuf  = xq + (size_t)B_ * S_ * E_;
    int8_t* hq    = gbuf + (size_t)B_ * G_;

    k_init  <<<1,    64,  0, stream>>>(hdr);
    k_maxabs<<<512,  256, 0, stream>>>(emb,  V_ * E_, hdr + 0);
    k_maxabs<<<256,  256, 0, stream>>>(w_ih, G_ * E_, hdr + 1);
    k_maxabs<<<256,  256, 0, stream>>>(w_hh, G_ * H_, hdr + 2);
    k_maxabs<<<1,    256, 0, stream>>>(fc_w, O_ * H_, hdr + 3);
    k_quantw<<<256,  256, 0, stream>>>(w_ih, wih_q, G_ * E_, hdr + 1);
    k_quantw<<<256,  256, 0, stream>>>(w_hh, whh_q, G_ * H_, hdr + 2);
    k_bias  <<<8,    256, 0, stream>>>(b_ih, b_hh, bias);
    k_embq  <<<1024, 256, 0, stream>>>(text, emb, xq, hdr + 0);
    k_lstm  <<<NWG_, 256, 0, stream>>>(xq, wih_q, whh_q, bias, hdr, gbuf, hq, hdr + 4);
    k_fc    <<<1,    128, 0, stream>>>(hq, fc_w, hdr, out);
}
